// SimpleMinGRU_15977278341678
// MI455X (gfx1250) — matrix operand order repaired and checked by reading, not compiled, not run on hardware
//
#include <hip/hip_runtime.h>
#include <hip/hip_bf16.h>

typedef _Float16 v16h __attribute__((ext_vector_type(16)));
typedef _Float16 v8h  __attribute__((ext_vector_type(8)));
typedef float    v8f  __attribute__((ext_vector_type(8)));

#define CAT16(lo, hi) __builtin_shufflevector(lo, hi, 0,1,2,3,4,5,6,7,8,9,10,11,12,13,14,15)

constexpr int Bk = 128;
constexpr int Lk = 2048;
constexpr int Hk = 256;
constexpr int ROWS = Bk * Lk;          // 262144 rows of the flattened (B*L, H) activations

// ---------------------------------------------------------------------------
// Kernel 0: convert Wz / Wh (f32, K x N row-major) -> f16 transposed (N x K)
// so each WMMA B-fragment lane reads a contiguous run of K values.
// ---------------------------------------------------------------------------
__global__ __launch_bounds__(256) void mingru_prep_weights(
    const float* __restrict__ Wz, const float* __restrict__ Wh,
    _Float16* __restrict__ WzT, _Float16* __restrict__ WhT) {
  int idx = blockIdx.x * 256 + threadIdx.x;     // 0 .. H*H-1
  if (idx >= Hk * Hk) return;
  int k = idx / Hk;
  int n = idx % Hk;
  WzT[n * Hk + k] = (_Float16)Wz[idx];
  WhT[n * Hk + k] = (_Float16)Wh[idx];
}

// ---------------------------------------------------------------------------
// Kernel 1: inp = x @ Wp + bp   (K=3), emitted as f16 row-major (B*L, H)
// ---------------------------------------------------------------------------
__global__ __launch_bounds__(256) void mingru_proj(
    const float* __restrict__ x, const float* __restrict__ Wp,
    const float* __restrict__ bp, _Float16* __restrict__ inp_h) {
  int row = blockIdx.x;                 // (b*L + l)
  int h   = threadIdx.x;                // channel
  const float* xr = x + (size_t)row * 3;
  float v = xr[0] * Wp[h] + xr[1] * Wp[Hk + h] + xr[2] * Wp[2 * Hk + h] + bp[h];
  inp_h[(size_t)row * Hk + h] = (_Float16)v;
}

// ---------------------------------------------------------------------------
// Kernel 2: z = sigmoid(inp@Wz+bz), h~ = tanh(inp@Wh+bh) via v_wmma_f32_16x16x32_f16.
// One wave per 16x16 output tile; both gate matrices share the A fragment.
// ---------------------------------------------------------------------------
__global__ __launch_bounds__(256) void mingru_gates(
    const _Float16* __restrict__ A,      // (ROWS, H) f16
    const _Float16* __restrict__ WzT,    // (N, K) f16
    const _Float16* __restrict__ WhT,    // (N, K) f16
    const float* __restrict__ bz, const float* __restrict__ bh,
    _Float16* __restrict__ Zo, _Float16* __restrict__ Ho) {
  int wave = threadIdx.x >> 5;
  int lane = threadIdx.x & 31;
  unsigned gw = blockIdx.x * 8u + (unsigned)wave;   // global wave id
  int tileN = gw & 15;                              // 16 N-tiles
  int tileM = gw >> 4;                              // 16384 M-tiles
  int m    = lane & 15;
  int half = lane >> 4;

  const _Float16* Arow = A + (size_t)(tileM * 16 + m) * Hk;
  const _Float16* Bzl  = WzT + (size_t)(tileN * 16 + m) * Hk + half * 16;
  const _Float16* Bhl  = WhT + (size_t)(tileN * 16 + m) * Hk + half * 16;

  v8f accz = {};
  v8f acch = {};
#pragma unroll
  for (int kk = 0; kk < 8; ++kk) {
    int k0 = kk * 32;
    // A fragment: lanes 0-15 row m K [0..15], lanes 16-31 K [16..31]
    v8h alo = *(const v8h*)(Arow + k0 + half * 16);
    v8h ahi = *(const v8h*)(Arow + k0 + half * 16 + 8);
    v16h a = CAT16(alo, ahi);
    // B fragments: lane = column n, half selects K 0..15 vs 16..31, contiguous in WT
    v8h bzlo = *(const v8h*)(Bzl + k0);
    v8h bzhi = *(const v8h*)(Bzl + k0 + 8);
    v16h bfz = CAT16(bzlo, bzhi);
    v8h bhlo = *(const v8h*)(Bhl + k0);
    v8h bhhi = *(const v8h*)(Bhl + k0 + 8);
    v16h bfh = CAT16(bhlo, bhhi);

    accz = __builtin_amdgcn_wmma_f32_16x16x32_f16(false, a, false, bfz,
                                                  (short)0, accz, false, false);
    acch = __builtin_amdgcn_wmma_f32_16x16x32_f16(false, a, false, bfh,
                                                  (short)0, acch, false, false);
  }

  int col = tileN * 16 + m;
  float biasz = bz[col];
  float biash = bh[col];
  size_t rowbase = (size_t)tileM * 16 + (size_t)half * 8;
#pragma unroll
  for (int r = 0; r < 8; ++r) {
    size_t off = (rowbase + r) * Hk + col;
    float vz = accz[r] + biasz;
    float vh = acch[r] + biash;
    Zo[off] = (_Float16)(1.0f / (1.0f + __expf(-vz)));
    Ho[off] = (_Float16)tanhf(vh);
  }
}

// ---------------------------------------------------------------------------
// Kernel 3: sequential scan. One block per batch, thread = channel h.
//   pred_t = h_{t-1} . Wg + bg   (emitted BEFORE the state update)
//   h_t    = (1-z_t) h_{t-1} + z_t h~_t
// Per-t reduction: wave32 shfl tree -> 8 partials staged in LDS per 256-step
// chunk, summed cooperatively (2 barriers per chunk instead of per step).
// ---------------------------------------------------------------------------
__global__ __launch_bounds__(256) void mingru_scan(
    const _Float16* __restrict__ Z, const _Float16* __restrict__ Ht,
    const float* __restrict__ Wg, const float* __restrict__ bg,
    float* __restrict__ out) {
  __shared__ float part[8 * 256];
  int b    = blockIdx.x;
  int h    = threadIdx.x;
  int wave = h >> 5;
  int lane = h & 31;

  float wg  = Wg[h];
  float bgv = bg[0];
  float hs  = 0.0f;

  const _Float16* zrow = Z  + (size_t)b * Lk * Hk + h;
  const _Float16* grow = Ht + (size_t)b * Lk * Hk + h;

  for (int tc = 0; tc < Lk; tc += 256) {
    for (int tt = 0; tt < 256; ++tt) {
      int t = tc + tt;
      // prediction contribution from h_{t-1}
      float c = hs * wg;
      c += __shfl_xor(c, 16, 32);
      c += __shfl_xor(c, 8, 32);
      c += __shfl_xor(c, 4, 32);
      c += __shfl_xor(c, 2, 32);
      c += __shfl_xor(c, 1, 32);
      if (lane == 0) part[wave * 256 + tt] = c;
      // state update
      float zt = (float)zrow[(size_t)t * Hk];
      float gt = (float)grow[(size_t)t * Hk];
      hs = (1.0f - zt) * hs + zt * gt;
    }
    __syncthreads();
    float s = bgv;
#pragma unroll
    for (int w = 0; w < 8; ++w) s += part[w * 256 + h];
    out[(size_t)b * Lk + tc + h] = s;
    __syncthreads();
  }
}

// ---------------------------------------------------------------------------
// Host-side launcher
// ---------------------------------------------------------------------------
extern "C" void kernel_launch(void* const* d_in, const int* in_sizes, int n_in,
                              void* d_out, int out_size, void* d_ws, size_t ws_size,
                              hipStream_t stream) {
  (void)in_sizes; (void)n_in; (void)out_size; (void)ws_size;
  const float* x  = (const float*)d_in[0];
  const float* Wp = (const float*)d_in[1];
  const float* bp = (const float*)d_in[2];
  const float* Wz = (const float*)d_in[3];
  const float* bz = (const float*)d_in[4];
  const float* Wh = (const float*)d_in[5];
  const float* bh = (const float*)d_in[6];
  const float* Wg = (const float*)d_in[7];
  const float* bg = (const float*)d_in[8];
  float* out = (float*)d_out;

  char* ws = (char*)d_ws;
  _Float16* inp_h = (_Float16*)(ws);                              // 128 MB
  _Float16* z16   = (_Float16*)(ws + 134217728ull);               // 128 MB
  _Float16* ht16  = (_Float16*)(ws + 268435456ull);               // 128 MB
  _Float16* WzT   = (_Float16*)(ws + 402653184ull);               // 128 KB
  _Float16* WhT   = (_Float16*)(ws + 402653184ull + 131072ull);   // 128 KB

  // 0) weight convert+transpose
  mingru_prep_weights<<<(Hk * Hk + 255) / 256, 256, 0, stream>>>(Wz, Wh, WzT, WhT);
  // 1) input projection -> f16
  mingru_proj<<<ROWS, 256, 0, stream>>>(x, Wp, bp, inp_h);
  // 2) WMMA gate GEMMs (+ activations)
  int nWaves = (ROWS / 16) * 16;          // Mtiles * Ntiles
  mingru_gates<<<nWaves / 8, 256, 0, stream>>>(inp_h, WzT, WhT, bz, bh, z16, ht16);
  // 3) sequential recurrence + gauss head
  mingru_scan<<<Bk, 256, 0, stream>>>(z16, ht16, Wg, bg, out);
}
